// QKVAttention_89421219103040
// MI455X (gfx1250) — matrix operand order repaired and checked by reading, not compiled, not run on hardware
//
#include <hip/hip_runtime.h>

typedef __attribute__((ext_vector_type(16))) _Float16 v16h;
typedef __attribute__((ext_vector_type(8)))  _Float16 v8h;
typedef __attribute__((ext_vector_type(2)))  _Float16 v2h;
typedef __attribute__((ext_vector_type(8)))  float    v8f;
typedef __attribute__((ext_vector_type(4)))  float    v4f;
typedef __attribute__((ext_vector_type(2)))  float    v2f;

namespace {
constexpr int kHeads = 64;   // bs * N_HEADS = 4 * 16
constexpr int kCh    = 64;   // head dim
constexpr int kT     = 2048; // sequence length
constexpr int kWaves = 8;    // waves per attention workgroup (share K/V via WGP$)
constexpr int kPStride = 72; // P staging row stride in halfs (144B = 9*16B)
constexpr int kOStride = 20; // O staging row stride in floats (80B = 5*16B)
// Per-side Q/K scale: 64^(-1/4) * sqrt(log2(e)) -> scores arrive in exp2 domain.
constexpr float kScaleQK = 0.42466090744206675f;
}

// ---------------------------------------------------------------------------
// Pass 1: pack qkv (fp32, per head: q[c][t], k[c][s], v[c][s]) into f16 ws:
//   qT[head][t][c] = q[c][t] * kScaleQK   (transposed)
//   kT[head][s][c] = k[c][s] * kScaleQK   (transposed)
//   vh[head][c][s] = v[c][s]              (same layout)
// ---------------------------------------------------------------------------
__global__ __launch_bounds__(256)
void qkv_pack_kernel(const float* __restrict__ qkv, _Float16* __restrict__ ws)
{
  __shared__ float tile[kCh][kCh + 1];
  const int tid  = threadIdx.x;
  const int head = blockIdx.x;
  const int t0   = blockIdx.y * kCh;      // 64-wide t tile

  const float* __restrict__ q = qkv + (size_t)head * 3 * kCh * kT;
  const float* __restrict__ k = q + (size_t)kCh * kT;
  const float* __restrict__ v = k + (size_t)kCh * kT;

  _Float16* __restrict__ qT  = ws + (size_t)head * kT * kCh;
  _Float16* __restrict__ kT2 = ws + (size_t)kHeads * kT * kCh + (size_t)head * kT * kCh;
  _Float16* __restrict__ vh  = ws + (size_t)2 * kHeads * kT * kCh + (size_t)head * kCh * kT;

  for (int i = tid; i < kCh * kCh; i += 256) {
    const int c = i >> 6, tt = i & 63;
    tile[c][tt] = q[(size_t)c * kT + t0 + tt];
  }
  __syncthreads();
  for (int i = tid; i < (kCh * kCh) / 2; i += 256) {
    const int tt = i >> 5, c = (i & 31) * 2;
    v2h p; p[0] = (_Float16)(tile[c][tt] * kScaleQK);
           p[1] = (_Float16)(tile[c + 1][tt] * kScaleQK);
    *(v2h*)(qT + (size_t)(t0 + tt) * kCh + c) = p;
  }
  __syncthreads();
  for (int i = tid; i < kCh * kCh; i += 256) {
    const int c = i >> 6, tt = i & 63;
    tile[c][tt] = k[(size_t)c * kT + t0 + tt];
  }
  __syncthreads();
  for (int i = tid; i < (kCh * kCh) / 2; i += 256) {
    const int tt = i >> 5, c = (i & 31) * 2;
    v2h p; p[0] = (_Float16)(tile[c][tt] * kScaleQK);
           p[1] = (_Float16)(tile[c + 1][tt] * kScaleQK);
    *(v2h*)(kT2 + (size_t)(t0 + tt) * kCh + c) = p;
  }
  for (int i = tid; i < (kCh * kCh) / 2; i += 256) {
    const int c = i >> 5, tt2 = (i & 31) * 2;
    const v2f w = *(const v2f*)(v + (size_t)c * kT + t0 + tt2);
    v2h p; p[0] = (_Float16)w[0]; p[1] = (_Float16)w[1];
    *(v2h*)(vh + (size_t)c * kT + t0 + tt2) = p;
  }
}

static __device__ __forceinline__ v16h cat8(v8h lo, v8h hi_)
{
  v16h r;
  #pragma unroll
  for (int j = 0; j < 8; ++j) { r[j] = lo[j]; r[8 + j] = hi_[j]; }
  return r;
}

// Max-reduce across each 16-lane row using DPP only (no LDS):
// xor1 (quad_perm [1,0,3,2]), xor2 (quad_perm [2,3,0,1]),
// ROW_MIRROR (0x140), ROW_HALF_MIRROR (0x141).
static __device__ __forceinline__ float rowmax16(float x)
{
  int t;
  t = __builtin_amdgcn_update_dpp(0, __float_as_int(x), 0xB1, 0xf, 0xf, true);
  x = fmaxf(x, __int_as_float(t));
  t = __builtin_amdgcn_update_dpp(0, __float_as_int(x), 0x4E, 0xf, 0xf, true);
  x = fmaxf(x, __int_as_float(t));
  t = __builtin_amdgcn_update_dpp(0, __float_as_int(x), 0x140, 0xf, 0xf, true);
  x = fmaxf(x, __int_as_float(t));
  t = __builtin_amdgcn_update_dpp(0, __float_as_int(x), 0x141, 0xf, 0xf, true);
  x = fmaxf(x, __int_as_float(t));
  return x;
}

// ---------------------------------------------------------------------------
// Pass 2: flash attention, f16 WMMA, fp32 accumulators, exp2-domain softmax.
// One wave owns 16 queries of one head; keys processed in blocks of 64.
// 8 waves per block walk identical K/V addresses -> WGP$ dedups L2 traffic.
// ---------------------------------------------------------------------------
__global__ __launch_bounds__(kWaves * 32)
void qkv_attention_fa_kernel(const _Float16* __restrict__ ws,
                             float* __restrict__ out)
{
  const int lane = threadIdx.x & 31;
  const int wave = threadIdx.x >> 5;
  const int ln   = lane & 15;
  const int hi   = lane >> 4;          // 0: lanes 0-15, 1: lanes 16-31

  const int head   = blockIdx.x;
  const int t_base = (blockIdx.y * kWaves + wave) * 16;

  const _Float16* __restrict__ qT  = ws + (size_t)head * kT * kCh;
  const _Float16* __restrict__ kT2 = ws + (size_t)kHeads * kT * kCh + (size_t)head * kT * kCh;
  const _Float16* __restrict__ vh  = ws + (size_t)2 * kHeads * kT * kCh + (size_t)head * kCh * kT;

  // ---- Q A-fragments (M=t, K=c): lane row t=lane&15,
  // half h -> c = h + 16*hi  => chunks {2*hi, 2*hi+1} and {4+2*hi, 5+2*hi}.
  const v8h* qrow = (const v8h*)(qT + (size_t)(t_base + ln) * kCh);
  const v16h aq0 = cat8(qrow[2 * hi],     qrow[2 * hi + 1]);
  const v16h aq1 = cat8(qrow[4 + 2 * hi], qrow[5 + 2 * hi]);

  v16h ones;
  #pragma unroll
  for (int j = 0; j < 16; ++j) ones[j] = (_Float16)1.0f;

  __shared__ _Float16 p_lds[kWaves][16][kPStride];
  __shared__ float    o_lds[kWaves][kCh][kOStride];
  _Float16 (*pbuf)[kPStride] = p_lds[wave];
  float    (*obuf)[kOStride] = o_lds[wave];

  float m_run[8];
  v8f o[4] = {v8f{}, v8f{}, v8f{}, v8f{}};   // O tile: M=t, N=c subtiles
  v8f lacc = {};                             // softmax denominators
  #pragma unroll
  for (int r = 0; r < 8; ++r) m_run[r] = -1e30f;

  for (int s_base = 0; s_base < kT; s_base += 64) {
    // ---- Scores S[16t x 64s]: 4 D-tiles, 2 chained WMMAs each (K = 64 c).
    v8f sc[4];
    #pragma unroll
    for (int ss = 0; ss < 4; ++ss) {
      const v8h* kp = (const v8h*)(kT2 + (size_t)(s_base + ss * 16 + ln) * kCh + hi * 16);
      v8f acc = {};
      acc = __builtin_amdgcn_wmma_f32_16x16x32_f16(false, aq0, false,
              cat8(kp[0], kp[1]), (short)0, acc, false, false);
      acc = __builtin_amdgcn_wmma_f32_16x16x32_f16(false, aq1, false,
              cat8(kp[4], kp[5]), (short)0, acc, false, false);
      sc[ss] = acc;
    }

    // ---- Online softmax (exp2 domain), DPP-only row max.
    #pragma unroll
    for (int r = 0; r < 8; ++r) {
      float rm = fmaxf(fmaxf(sc[0][r], sc[1][r]), fmaxf(sc[2][r], sc[3][r]));
      rm = rowmax16(rm);
      const float m_new = fmaxf(m_run[r], rm);
      const float alpha = exp2f(m_run[r] - m_new);
      m_run[r] = m_new;
      const float p0 = exp2f(sc[0][r] - m_new);
      const float p1 = exp2f(sc[1][r] - m_new);
      const float p2 = exp2f(sc[2][r] - m_new);
      const float p3 = exp2f(sc[3][r] - m_new);
      lacc[r] *= alpha;
      o[0][r] *= alpha; o[1][r] *= alpha; o[2][r] *= alpha; o[3][r] *= alpha;
      const int tl = r + hi * 8;
      pbuf[tl][ln]      = (_Float16)p0;
      pbuf[tl][ln + 16] = (_Float16)p1;
      pbuf[tl][ln + 32] = (_Float16)p2;
      pbuf[tl][ln + 48] = (_Float16)p3;
    }

    asm volatile("s_wait_dscnt 0" ::: "memory");  // LDS RAW within wave

    // ---- P as A-fragments (b128 reads, conflict-free padded rows).
    const v8h* prow = (const v8h*)(&pbuf[ln][0]);
    const v16h ap0 = cat8(prow[2 * hi],     prow[2 * hi + 1]);   // s 0..31
    const v16h ap1 = cat8(prow[4 + 2 * hi], prow[5 + 2 * hi]);   // s 32..63

    // ---- Row sums via ones-WMMA (every D column equals the row sum).
    lacc = __builtin_amdgcn_wmma_f32_16x16x32_f16(false, ap0, false, ones,
                                                  (short)0, lacc, false, false);
    lacc = __builtin_amdgcn_wmma_f32_16x16x32_f16(false, ap1, false, ones,
                                                  (short)0, lacc, false, false);

    // ---- O += P * V^T: V B-frags are 16 contiguous halfs per lane.
    #pragma unroll
    for (int ct = 0; ct < 4; ++ct) {
      const v8h* vp = (const v8h*)(vh + (size_t)(ct * 16 + ln) * kT + s_base + hi * 16);
      o[ct] = __builtin_amdgcn_wmma_f32_16x16x32_f16(false, ap0, false,
                cat8(vp[0], vp[1]), (short)0, o[ct], false, false);
      o[ct] = __builtin_amdgcn_wmma_f32_16x16x32_f16(false, ap1, false,
                cat8(vp[4], vp[5]), (short)0, o[ct], false, false);
    }
  }

  // ---- Epilogue: normalize, transpose through LDS, store b128-contiguous.
  #pragma unroll
  for (int r = 0; r < 8; ++r) {
    const float inv_l = 1.0f / lacc[r];
    const int tl = r + hi * 8;
    #pragma unroll
    for (int ct = 0; ct < 4; ++ct)
      obuf[ct * 16 + ln][tl] = o[ct][r] * inv_l;
  }

  asm volatile("s_wait_dscnt 0" ::: "memory");  // LDS RAW within wave

  #pragma unroll
  for (int half = 0; half < 2; ++half) {
    const int c = lane + half * 32;
    const v4f* src = (const v4f*)(&obuf[c][0]);
    v4f* dst = (v4f*)(out + ((size_t)head * kCh + c) * kT + t_base);
    #pragma unroll
    for (int j = 0; j < 4; ++j) dst[j] = src[j];
  }
}

extern "C" void kernel_launch(void* const* d_in, const int* in_sizes, int n_in,
                              void* d_out, int out_size, void* d_ws, size_t ws_size,
                              hipStream_t stream) {
  (void)in_sizes; (void)n_in; (void)out_size; (void)ws_size;
  const float* qkv = (const float*)d_in[0];
  float* out = (float*)d_out;
  _Float16* ws = (_Float16*)d_ws;   // needs 3*64*2048*64*2 = 50.3 MB

  dim3 pgrid(kHeads, kT / kCh);     // 64 heads x 32 t-tiles
  qkv_pack_kernel<<<pgrid, 256, 0, stream>>>(qkv, ws);

  dim3 agrid(kHeads, (kT / 16) / kWaves);  // 64 heads x 16 blocks of 8 q-tiles
  qkv_attention_fa_kernel<<<agrid, kWaves * 32, 0, stream>>>(ws, out);
}
